// MultiBoxLoss_31568009625820
// MI455X (gfx1250) — compile-verified, matrix operand order checked by reading, not run on hardware
//
#include <hip/hip_runtime.h>
#include <hip/hip_bf16.h>

// ---------------- constants from the reference ----------------
#define NCLS   81
#define BB     64
#define PP     8732
#define MM     24
#define THRESH 0.5f
#define NEGPOS 3
#define TPB    256

typedef _Float16 v16h __attribute__((ext_vector_type(16)));
typedef float    v8f  __attribute__((ext_vector_type(8)));

// ---------------- deterministic block reductions (blockDim.x == 256) ------
__device__ __forceinline__ float blockReduceSumF(float v, float* sm) {
  int t = threadIdx.x;
  sm[t] = v; __syncthreads();
  #pragma unroll
  for (int s = 128; s > 0; s >>= 1) {
    if (t < s) sm[t] += sm[t + s];
    __syncthreads();
  }
  float r = sm[0]; __syncthreads();
  return r;
}
__device__ __forceinline__ int blockReduceSumI(int v, int* sm) {
  int t = threadIdx.x;
  sm[t] = v; __syncthreads();
  #pragma unroll
  for (int s = 128; s > 0; s >>= 1) {
    if (t < s) sm[t] += sm[t + s];
    __syncthreads();
  }
  int r = sm[0]; __syncthreads();
  return r;
}

// ---------------- K0: init best-prior argmax accumulators ----------------
__global__ void mb_k0_init(unsigned long long* __restrict__ bestPrior) {
  int i = blockIdx.x * blockDim.x + threadIdx.x;
  if (i < BB * MM) bestPrior[i] = 0ull;
}

// ---------------- K1a: IoU match ----------------
// per (b,p): best truth (val,idx); per (b,m): best prior via u64 atomicMax
// key = float_bits(iou)<<32 | (0xFFFFFFFF - p)  -> max picks max iou, then min p
__global__ void mb_k1a_match(const float* __restrict__ priors,
                             const float* __restrict__ truths,
                             float* __restrict__ btVal,
                             int*   __restrict__ btIdx,
                             unsigned long long* __restrict__ bestPrior) {
  __shared__ float sT[MM * 4];
  __shared__ float sArea[MM];
  __shared__ unsigned long long sCand[MM];
  const int b = blockIdx.y;
  const int t = threadIdx.x;
  if (t < MM * 4) sT[t] = truths[b * MM * 4 + t];
  if (t < MM)     sCand[t] = 0ull;
  __syncthreads();
  if (t < MM)
    sArea[t] = (sT[t*4+2] - sT[t*4+0]) * (sT[t*4+3] - sT[t*4+1]);
  __syncthreads();

  const int p = blockIdx.x * blockDim.x + t;
  if (p < PP) {
    float4 pr = ((const float4*)priors)[p];
    const float px1 = pr.x - pr.z * 0.5f, py1 = pr.y - pr.w * 0.5f;
    const float px2 = pr.x + pr.z * 0.5f, py2 = pr.y + pr.w * 0.5f;
    const float areaP = (px2 - px1) * (py2 - py1);
    float bv = -1.0f; int bi = 0;
    #pragma unroll
    for (int m = 0; m < MM; ++m) {
      float ix = fminf(sT[m*4+2], px2) - fmaxf(sT[m*4+0], px1); ix = fmaxf(ix, 0.f);
      float iy = fminf(sT[m*4+3], py2) - fmaxf(sT[m*4+1], py1); iy = fmaxf(iy, 0.f);
      float inter = ix * iy;
      float iou = inter / (sArea[m] + areaP - inter);
      if (iou > bv) { bv = iou; bi = m; }   // first-index tie-break (ascending m)
      unsigned long long key =
          ((unsigned long long)__float_as_uint(iou) << 32) |
          (unsigned long long)(0xFFFFFFFFu - (unsigned)p);
      atomicMax(&sCand[m], key);            // ds_max_u64, order-independent
    }
    btVal[b * PP + p] = bv;
    btIdx[b * PP + p] = bi;
  }
  __syncthreads();
  if (t < MM) atomicMax(&bestPrior[b * MM + t], sCand[t]);  // global_atomic_max_u64
}

// ---------------- K1b: override + conf_t + encode + smooth L1 ----------------
__global__ void mb_k1b_encode(const float* __restrict__ loc_data,
                              const float* __restrict__ priors,
                              const float* __restrict__ truths,
                              const int*   __restrict__ labels,
                              const float* __restrict__ btVal,
                              const int*   __restrict__ btIdx,
                              const unsigned long long* __restrict__ bestPrior,
                              int*   __restrict__ confT,
                              float* __restrict__ lossLpart,
                              int*   __restrict__ posPart) {
  __shared__ float    sT[MM * 4];
  __shared__ int      sL[MM];
  __shared__ unsigned sBP[MM];
  __shared__ float fred[TPB];
  __shared__ int   ired[TPB];
  const int b = blockIdx.y;
  const int t = threadIdx.x;
  if (t < MM * 4) sT[t] = truths[b * MM * 4 + t];
  if (t < MM) {
    sL[t] = labels[b * MM + t];
    unsigned long long k = bestPrior[b * MM + t];
    sBP[t] = 0xFFFFFFFFu - (unsigned)(k & 0xFFFFFFFFull);  // decoded best prior p*
  }
  __syncthreads();

  const int p = blockIdx.x * blockDim.x + t;
  float ll = 0.f; int pos = 0;
  if (p < PP) {
    const int r = b * PP + p;
    float val = btVal[r]; int idx = btIdx[r];
    #pragma unroll
    for (int m = 0; m < MM; ++m)
      if (sBP[m] == (unsigned)p) { idx = m; val = 2.0f; }  // last m wins (JAX scatter)
    int conf = sL[idx] + 1;
    if (val < THRESH) conf = 0;
    confT[r] = conf;
    if (conf > 0) {
      float4 pr = ((const float4*)priors)[p];
      const float m0 = sT[idx*4+0], m1 = sT[idx*4+1];
      const float m2 = sT[idx*4+2], m3 = sT[idx*4+3];
      const float g0 = ((m0 + m2) * 0.5f - pr.x) / (0.1f * pr.z);
      const float g1 = ((m1 + m3) * 0.5f - pr.y) / (0.1f * pr.w);
      const float g2 = logf((m2 - m0) / pr.z) / 0.2f;
      const float g3 = logf((m3 - m1) / pr.w) / 0.2f;
      float4 ld = ((const float4*)loc_data)[r];
      const float d[4] = { ld.x - g0, ld.y - g1, ld.z - g2, ld.w - g3 };
      #pragma unroll
      for (int i = 0; i < 4; ++i) {
        float ad = fabsf(d[i]);
        ll += (ad < 1.f) ? 0.5f * d[i] * d[i] : ad - 0.5f;
      }
      pos = 1;
    }
  }
  float s = blockReduceSumF(ll, fred);
  int   c = blockReduceSumI(pos, ired);
  if (t == 0) {
    int blk = b * gridDim.x + blockIdx.x;
    lossLpart[blk] = s;
    posPart[blk]  = c;
  }
}

// ---------------- K2: logsumexp cross-entropy via WMMA row-sums ----------------
// One wave = 16 contiguous rows (row = b*PP+p flattened; 558848 = 16*34928).
// Tile staged in LDS (coalesced), row-sum of exp(x-max) computed with
// v_wmma_f32_16x16x32_f16 against an all-ones B, using hi/lo f16 split for
// f32-accurate accumulation. Row-sums are K-permutation invariant, so only
// the lane->M mapping matters (lanes 0-15 and 16-31 both carry M=0..15).
__global__ void mb_k2_ce(const float* __restrict__ conf_data,
                         const int*   __restrict__ confT,
                         float*       __restrict__ ceAll) {
  __shared__ float sTile[8][16 * 96];   // 8 waves x (16 rows x 96 padded classes)
  __shared__ float sRs[8][16];
  const int tid = threadIdx.x;
  const int w = tid >> 5, l = tid & 31;
  const int tile = blockIdx.x * 8 + w;
  const long long row0 = (long long)tile * 16;
  const float* base = conf_data + row0 * NCLS;
  float* T = sTile[w];

  // 1) coalesced fill: tile is one contiguous 1296-float block
  for (int e = l; e < 16 * NCLS; e += 32) {
    int rr = e / NCLS;
    int cc = e - rr * NCLS;
    T[rr * 96 + cc] = base[e];
  }
  // 2) row max: lane (r,h) scans half the row, combine across halves
  const int r = l & 15, h = l >> 4;
  float mx = -1e30f;
  const int c0m = h ? 41 : 0, c1m = h ? NCLS : 41;
  for (int c = c0m; c < c1m; ++c) mx = fmaxf(mx, T[r * 96 + c]);
  mx = fmaxf(mx, __shfl_xor(mx, 16));
  // 3) exp(x - max), zero-pad classes 81..95
  for (int c = h * 48; c < h * 48 + 48; ++c) {
    float v = (c < NCLS) ? expf(T[r * 96 + c] - mx) : 0.f;
    T[r * 96 + c] = v;
  }
  // 4) WMMA row sums: 3 chunks of K=32, hi+lo split => 6 WMMAs, f32 acc
  v16h ones;
  #pragma unroll
  for (int i = 0; i < 16; ++i) ones[i] = (_Float16)1.0f;
  v8f acc;
  #pragma unroll
  for (int i = 0; i < 8; ++i) acc[i] = 0.f;

  #pragma unroll
  for (int c0 = 0; c0 < 96; c0 += 32) {
    // per ISA A-layout: lane (r,h) holds K = c0+8h+[0..7] and c0+16+8h+[0..7]
    const float4* q1 = (const float4*)&T[r * 96 + c0 + 8 * h];
    const float4* q2 = (const float4*)&T[r * 96 + c0 + 16 + 8 * h];
    float4 a0 = q1[0], a1 = q1[1], a2 = q2[0], a3 = q2[1];
    float xs[16] = { a0.x, a0.y, a0.z, a0.w, a1.x, a1.y, a1.z, a1.w,
                     a2.x, a2.y, a2.z, a2.w, a3.x, a3.y, a3.z, a3.w };
    v16h ah, al;
    #pragma unroll
    for (int i = 0; i < 16; ++i) {
      _Float16 hi = (_Float16)xs[i];
      ah[i] = hi;
      al[i] = (_Float16)(xs[i] - (float)hi);
    }
    acc = __builtin_amdgcn_wmma_f32_16x16x32_f16(false, ah, false, ones,
                                                 (short)0, acc, false, false);
    acc = __builtin_amdgcn_wmma_f32_16x16x32_f16(false, al, false, ones,
                                                 (short)0, acc, false, false);
  }
  // 5) D layout: VGPR j => M=j (lanes 0-15) / M=j+8 (lanes 16-31); row-sum
  //    replicated over N, so lane 0 holds rows 0..7, lane 16 rows 8..15.
  if (l == 0) {
    #pragma unroll
    for (int j = 0; j < 8; ++j) sRs[w][j] = acc[j];
  }
  if (l == 16) {
    #pragma unroll
    for (int j = 0; j < 8; ++j) sRs[w][8 + j] = acc[j];
  }
  // DS ops are in-order within a wave: safe to read back
  if (l < 16) {
    long long row = row0 + l;
    float rs = sRs[w][l];
    int tc = confT[row];
    float picked = conf_data[row * NCLS + tc];
    ceAll[row] = mx + logf(rs) - picked;   // lse - picked, >= 0
  }
}

// ---------------- K3: exact hard-negative mining (radix select) ----------------
__global__ void mb_k3_mine(const float* __restrict__ ceAll,
                           const int*   __restrict__ confT,
                           float*       __restrict__ lossCpart) {
  __shared__ unsigned keys[PP];
  __shared__ float fred[TPB];
  __shared__ int   ired[TPB];
  const int b = blockIdx.x;
  const int t = threadIdx.x;
  float posSum = 0.f; int npos = 0;
  for (int p = t; p < PP; p += TPB) {
    float ce = ceAll[b * PP + p];
    int cf = confT[b * PP + p];
    if (cf > 0) { posSum += ce; npos++; keys[p] = 0u; }       // loss_rank = 0 for pos
    else keys[p] = __float_as_uint(fmaxf(ce, 0.f));            // >=0 => bits monotone
  }
  __syncthreads();
  npos   = blockReduceSumI(npos, ired);
  posSum = blockReduceSumF(posSum, fred);
  const int numNeg = min(NEGPOS * npos, PP - 1);

  float sumNeg = 0.f;
  if (numNeg > 0) {                       // uniform across block
    unsigned cur = 0u; int kk = numNeg;
    for (int bit = 31; bit >= 0; --bit) { // MSB radix select of kth largest
      unsigned test = cur | (1u << bit);
      unsigned hi = test >> bit;
      int cnt = 0;
      for (int p = t; p < PP; p += TPB)
        cnt += ((keys[p] >> bit) == hi) ? 1 : 0;
      cnt = blockReduceSumI(cnt, ired);
      if (cnt >= kk) cur = test; else kk -= cnt;
    }
    int cg = 0; float sg = 0.f;
    for (int p = t; p < PP; p += TPB) {
      unsigned k = keys[p];
      if (k > cur) { cg++; sg += __uint_as_float(k); }
    }
    cg = blockReduceSumI(cg, ired);
    sg = blockReduceSumF(sg, fred);
    sumNeg = sg + (float)(numNeg - cg) * __uint_as_float(cur); // tie-exact top-K sum
  }
  if (t == 0) lossCpart[b] = posSum + sumNeg;
}

// ---------------- K4: final deterministic reduction ----------------
__global__ void mb_k4_final(const float* __restrict__ lossLpart,
                            const int*   __restrict__ posPart, int nblk,
                            const float* __restrict__ lossCpart,
                            float* __restrict__ out) {
  __shared__ float fred[TPB];
  __shared__ int   ired[TPB];
  const int t = threadIdx.x;
  float sl = 0.f; int np = 0;
  for (int i = t; i < nblk; i += TPB) { sl += lossLpart[i]; np += posPart[i]; }
  float sc = 0.f;
  for (int i = t; i < BB; i += TPB) sc += lossCpart[i];
  sl = blockReduceSumF(sl, fred);
  np = blockReduceSumI(np, ired);
  sc = blockReduceSumF(sc, fred);
  if (t == 0) {
    float N = (float)np;
    if (N < 1.f) N = 1.f;
    out[0] = sl / N;
    out[1] = sc / N;
  }
}

// ---------------- launcher ----------------
extern "C" void kernel_launch(void* const* d_in, const int* in_sizes, int n_in,
                              void* d_out, int out_size, void* d_ws, size_t ws_size,
                              hipStream_t stream) {
  const float* loc    = (const float*)d_in[0];   // [B,P,4]
  const float* conf   = (const float*)d_in[1];   // [B,P,81]
  const float* priors = (const float*)d_in[2];   // [P,4]
  const float* truths = (const float*)d_in[3];   // [B,M,4]
  const int*   labels = (const int*)d_in[4];     // [B,M]
  float* out = (float*)d_out;

  char* ws = (char*)d_ws;
  size_t off = 0;
  auto alloc = [&](size_t bytes) -> void* {
    void* p = ws + off;
    off = (off + bytes + 255) & ~(size_t)255;
    return p;
  };
  const int NBX  = (PP + TPB - 1) / TPB;     // 35
  const int NBLK = BB * NBX;                 // 2240

  float* btVal = (float*)alloc(sizeof(float) * BB * PP);
  int*   btIdx = (int*)  alloc(sizeof(int)   * BB * PP);
  unsigned long long* bestPrior =
      (unsigned long long*)alloc(sizeof(unsigned long long) * BB * MM);
  int*   confT = (int*)  alloc(sizeof(int)   * BB * PP);
  float* ceAll = (float*)alloc(sizeof(float) * BB * PP);
  float* lossLpart = (float*)alloc(sizeof(float) * NBLK);
  int*   posPart   = (int*)  alloc(sizeof(int)   * NBLK);
  float* lossCpart = (float*)alloc(sizeof(float) * BB);

  mb_k0_init<<<(BB * MM + TPB - 1) / TPB, TPB, 0, stream>>>(bestPrior);
  mb_k1a_match<<<dim3(NBX, BB), TPB, 0, stream>>>(priors, truths, btVal, btIdx,
                                                  bestPrior);
  mb_k1b_encode<<<dim3(NBX, BB), TPB, 0, stream>>>(loc, priors, truths, labels,
                                                   btVal, btIdx, bestPrior,
                                                   confT, lossLpart, posPart);
  const int ntiles = (BB * PP) / 16;         // 34928, exact
  mb_k2_ce<<<ntiles / 8, TPB, 0, stream>>>(conf, confT, ceAll);
  mb_k3_mine<<<BB, TPB, 0, stream>>>(ceAll, confT, lossCpart);
  mb_k4_final<<<1, TPB, 0, stream>>>(lossLpart, posPart, NBLK, lossCpart, out);
}
